// ComformerConvEdgeLayer_32366873542775
// MI455X (gfx1250) — matrix operand order repaired and checked by reading, not compiled, not run on hardware
//
#include <hip/hip_runtime.h>
#include <hip/hip_bf16.h>

// ---------------------------------------------------------------------------
// ComformerConvEdgeLayer fused kernel for gfx1250 (MI455X), bf16 WMMA path.
//   E = 50000 edges, C = 128 channels, 3 lattice directions.
// Strategy: pre-convert all weights to bf16 transposed (WT[n][k]) in d_ws,
// then one fused kernel: 32 edges per 256-thread block, every GEMM done with
// v_wmma_f32_16x16x32_bf16 (f32 accumulate), all intermediates in LDS.
// ---------------------------------------------------------------------------

typedef __bf16 bf16_t;
typedef __attribute__((ext_vector_type(16))) __bf16 v16bf;
typedef __attribute__((ext_vector_type(8)))  __bf16 v8bf;
typedef __attribute__((ext_vector_type(8)))  float  v8f;

#define C128 128
#define BE   32          // edges per block

// ws layout (bf16 element offsets)
#define OFF_WQ   0
#define OFF_WK   16384
#define OFF_WV   32768
#define OFF_WKE  49152     // 3 x 16384
#define OFF_WVE  98304     // 3 x 16384
#define OFF_WE   147456
#define OFF_WK2  163840
#define OFF_WM2  180224
#define OFF_WC   196608
#define OFF_WK1  212992    // 384x128
#define OFF_WM1  262144    // 384x128
#define WS_BF16_ELEMS 311296

// --- weight convert + transpose: WT[n*K + k] = (bf16) W[k*N + n] -----------
__global__ void cvt_transpose_bf16(const float* __restrict__ W,
                                   bf16_t* __restrict__ WT, int K, int N) {
  int i = blockIdx.x * 256 + threadIdx.x;
  if (i >= K * N) return;
  int k = i / N, n = i - k * N;
  WT[n * K + k] = (bf16_t)W[i];
}

// --- WMMA fragment loaders -------------------------------------------------
// A (16x32 bf16, MxK): lane<16 -> row m=lane, K = {kk..kk+7, kk+16..kk+23}
//                      lane>=16 -> same row block, K shifted by 8.
__device__ __forceinline__ v16bf load_a(const bf16_t* p) {
  v8bf lo = *(const v8bf*)p;
  v8bf hi = *(const v8bf*)(p + 16);
  v16bf f;
#pragma unroll
  for (int j = 0; j < 8; ++j) { f[j] = lo[j]; f[8 + j] = hi[j]; }
  return f;
}
// B (32x16 bf16, KxN): lane<16 -> col n, K = {kk..kk+7, kk+16..kk+23}; lane>=16 -> K shifted by 8.
__device__ __forceinline__ v16bf load_b(const bf16_t* p) {
  v8bf lo = *(const v8bf*)p;
  v8bf hi = *(const v8bf*)(p + 16);
  v16bf f;
#pragma unroll
  for (int j = 0; j < 8; ++j) { f[j] = lo[j]; f[8 + j] = hi[j]; }
  return f;
}

// --- 32xK (LDS bf16, row-major) @ KxC128 (global bf16, transposed rows) ----
// 8 waves: wave w -> M-tile (w>>2), N-tiles (w&3) and (w&3)+4.
// Epilogue f(m, n, val) called per accumulator element (global in 32x128 tile).
template <int K, typename F>
__device__ __forceinline__ void gemm_tile(const bf16_t* __restrict__ A, int lda,
                                          const bf16_t* __restrict__ BT, F f) {
  const int wave = threadIdx.x >> 5;
  const int lane = threadIdx.x & 31;
  const int mt   = wave >> 2;      // 0..1
  const int nt   = wave & 3;       // 0..3 (+4 for second tile)
  const int half = lane >> 4;
  const int lm   = lane & 15;
  v8f acc0 = {};
  v8f acc1 = {};
  const bf16_t* ap  = A  + (mt * 16 + lm) * lda + half * 8;
  const bf16_t* bp0 = BT + (nt * 16 + lm) * K + half * 8;
  const bf16_t* bp1 = BT + ((nt + 4) * 16 + lm) * K + half * 8;
#pragma unroll
  for (int kk = 0; kk < K; kk += 32) {
    v16bf a  = load_a(ap + kk);
    v16bf b0 = load_b(bp0 + kk);
    v16bf b1 = load_b(bp1 + kk);
    acc0 = __builtin_amdgcn_wmma_f32_16x16x32_bf16(false, a, false, b0, (short)0, acc0, false, false);
    acc1 = __builtin_amdgcn_wmma_f32_16x16x32_bf16(false, a, false, b1, (short)0, acc1, false, false);
  }
  const int nA = nt * 16 + lm;
  const int nB = (nt + 4) * 16 + lm;
#pragma unroll
  for (int r = 0; r < 8; ++r) {
    int m = mt * 16 + 8 * half + r;   // C/D layout: VGPR r -> M=r (+8 for upper lanes)
    f(m, nA, acc0[r]);
    f(m, nB, acc1[r]);
  }
}

// sum across the 8 lanes sharing one row (tid groups of 8; 8 | 32 so intra-wave)
__device__ __forceinline__ float red8(float v) {
  v += __shfl_xor(v, 1, 32);
  v += __shfl_xor(v, 2, 32);
  v += __shfl_xor(v, 4, 32);
  return v;
}

__launch_bounds__(256)
__global__ void comformer_edge_kernel(
    const float* __restrict__ edge, const float* __restrict__ nlen,
    const float* __restrict__ nang,
    const float* __restrict__ bq, const float* __restrict__ bk,
    const float* __restrict__ bv, const float* __restrict__ bke,
    const float* __restrict__ bve, const float* __restrict__ bk1,
    const float* __restrict__ bk2, const float* __restrict__ bm1,
    const float* __restrict__ bm2, const float* __restrict__ bc,
    const float* __restrict__ g_att, const float* __restrict__ b_att,
    const float* __restrict__ g_bn, const float* __restrict__ b_bn,
    const bf16_t* __restrict__ wsb, float* __restrict__ out, int E) {
  __shared__ bf16_t sTmp[BE * 128];     //  8 KB: A-staging / MLP hidden / S(bf16)
  __shared__ float  sQ[BE * 128];       // 16 KB: q projection (lives whole kernel)
  __shared__ bf16_t sKcat[BE * 384];    // 24 KB: [kx | ky_t | exy_t]
  __shared__ bf16_t sMcat[BE * 384];    // 24 KB: [vx | vy_t | exy_t]
  __shared__ float  sK[BE * 128];       // 16 KB: k_t -> gate -> final pre-LN
  __shared__ float  sS[BE * 128];       // 16 KB: sum_t msg_t * gate_t

  const int tid = threadIdx.x;
  const int e0  = blockIdx.x * BE;

  const bf16_t* WqT  = wsb + OFF_WQ;
  const bf16_t* WkT  = wsb + OFF_WK;
  const bf16_t* WvT  = wsb + OFF_WV;
  const bf16_t* WkeT = wsb + OFF_WKE;
  const bf16_t* WveT = wsb + OFF_WVE;
  const bf16_t* WeT  = wsb + OFF_WE;
  const bf16_t* Wk2T = wsb + OFF_WK2;
  const bf16_t* Wm2T = wsb + OFF_WM2;
  const bf16_t* WcT  = wsb + OFF_WC;
  const bf16_t* Wk1T = wsb + OFF_WK1;
  const bf16_t* Wm1T = wsb + OFF_WM1;

  // stage edge block (zero-padded) + zero the direction accumulator
  for (int i = tid; i < BE * 128; i += 256) {
    int r = i >> 7, c = i & 127;
    int e = e0 + r;
    float v = (e < E) ? edge[e * 128 + c] : 0.f;
    sTmp[i] = (bf16_t)v;
    sS[i]   = 0.f;
  }
  __syncthreads();

  // q / kx / vx projections (all read sTmp = edge)
  gemm_tile<128>(sTmp, 128, WqT, [&](int m, int n, float v) {
    sQ[m * 128 + n] = v + bq[n];
  });
  gemm_tile<128>(sTmp, 128, WkT, [&](int m, int n, float v) {
    sKcat[m * 384 + n] = (bf16_t)(v + bk[n]);
  });
  gemm_tile<128>(sTmp, 128, WvT, [&](int m, int n, float v) {
    sMcat[m * 384 + n] = (bf16_t)(v + bv[n]);
  });
  __syncthreads();

  for (int t = 0; t < 3; ++t) {
    // stage edge_nei_len[:, t, :]
    for (int i = tid; i < BE * 128; i += 256) {
      int r = i >> 7, c = i & 127;
      int e = e0 + r;
      float v = (e < E) ? nlen[(e * 3 + t) * 128 + c] : 0.f;
      sTmp[i] = (bf16_t)v;
    }
    __syncthreads();
    gemm_tile<128>(sTmp, 128, WkeT + t * 16384, [&](int m, int n, float v) {
      sKcat[m * 384 + 128 + n] = (bf16_t)(v + bke[t * 128 + n]);
    });
    gemm_tile<128>(sTmp, 128, WveT + t * 16384, [&](int m, int n, float v) {
      sMcat[m * 384 + 128 + n] = (bf16_t)(v + bve[t * 128 + n]);
    });
    __syncthreads();

    // stage edge_nei_angle[:, t, :] ; exy = ang @ We (no bias), fed to both cats
    for (int i = tid; i < BE * 128; i += 256) {
      int r = i >> 7, c = i & 127;
      int e = e0 + r;
      float v = (e < E) ? nang[(e * 3 + t) * 128 + c] : 0.f;
      sTmp[i] = (bf16_t)v;
    }
    __syncthreads();
    gemm_tile<128>(sTmp, 128, WeT, [&](int m, int n, float v) {
      bf16_t b = (bf16_t)v;
      sKcat[m * 384 + 256 + n] = b;
      sMcat[m * 384 + 256 + n] = b;
    });
    __syncthreads();

    // key MLP: silu(kcat @ Wk1 + bk1) @ Wk2 + bk2 -> sK
    gemm_tile<384>(sKcat, 384, Wk1T, [&](int m, int n, float v) {
      float x = v + bk1[n];
      sTmp[m * 128 + n] = (bf16_t)(x / (1.f + __expf(-x)));   // silu
    });
    __syncthreads();
    gemm_tile<128>(sTmp, 128, Wk2T, [&](int m, int n, float v) {
      sK[m * 128 + n] = v + bk2[n];
    });
    __syncthreads();

    // gate pass: sK := sigmoid( LN(q * k / sqrt(C)) * g_att + b_att )
    {
      const int row = tid >> 3, seg = tid & 7;
      const int base = row * 128 + seg * 16;
      const float rs = 0.08838834764831845f;   // 1/sqrt(128)
      float av[16], s1 = 0.f, s2 = 0.f;
#pragma unroll
      for (int j = 0; j < 16; ++j) {
        float a = sQ[base + j] * sK[base + j] * rs;
        av[j] = a; s1 += a; s2 += a * a;
      }
      s1 = red8(s1); s2 = red8(s2);
      float mean = s1 * (1.f / 128.f);
      float var  = s2 * (1.f / 128.f) - mean * mean;
      float inv  = rsqrtf(var + 1e-5f);
#pragma unroll
      for (int j = 0; j < 16; ++j) {
        int ch = seg * 16 + j;
        float an = (av[j] - mean) * inv * g_att[ch] + b_att[ch];
        sK[base + j] = 1.f / (1.f + __expf(-an));
      }
    }
    __syncthreads();

    // msg MLP: silu(mcat @ Wm1 + bm1) @ Wm2 + bm2, gated, accumulated into sS
    gemm_tile<384>(sMcat, 384, Wm1T, [&](int m, int n, float v) {
      float x = v + bm1[n];
      sTmp[m * 128 + n] = (bf16_t)(x / (1.f + __expf(-x)));
    });
    __syncthreads();
    gemm_tile<128>(sTmp, 128, Wm2T, [&](int m, int n, float v) {
      sS[m * 128 + n] += (v + bm2[n]) * sK[m * 128 + n];
    });
    __syncthreads();
  }

  // (sum_t out_t) @ Wc + 3*bc   (linearity: sum before lin_concate)
  for (int i = tid; i < BE * 128; i += 256) sTmp[i] = (bf16_t)sS[i];
  __syncthreads();
  gemm_tile<128>(sTmp, 128, WcT, [&](int m, int n, float v) {
    sK[m * 128 + n] = v + 3.f * bc[n];
  });
  __syncthreads();

  // final: softplus( edge + LN(out) * g_bn + b_bn )
  {
    const int row = tid >> 3, seg = tid & 7;
    const int base = row * 128 + seg * 16;
    float s1 = 0.f, s2 = 0.f;
#pragma unroll
    for (int j = 0; j < 16; ++j) {
      float x = sK[base + j];
      s1 += x; s2 += x * x;
    }
    s1 = red8(s1); s2 = red8(s2);
    float mean = s1 * (1.f / 128.f);
    float var  = s2 * (1.f / 128.f) - mean * mean;
    float inv  = rsqrtf(var + 1e-5f);
    int e = e0 + row;
    if (e < E) {
#pragma unroll
      for (int j = 0; j < 16; ++j) {
        int ch = seg * 16 + j;
        float y = (sK[base + j] - mean) * inv * g_bn[ch] + b_bn[ch];
        float z = edge[e * 128 + ch] + y;
        float sp = (z > 0.f) ? z + log1pf(__expf(-z)) : log1pf(__expf(z));
        out[e * 128 + ch] = sp;
      }
    }
  }
}

extern "C" void kernel_launch(void* const* d_in, const int* in_sizes, int n_in,
                              void* d_out, int out_size, void* d_ws, size_t ws_size,
                              hipStream_t stream) {
  (void)n_in; (void)out_size; (void)ws_size;
  const float* edge = (const float*)d_in[0];
  const float* nlen = (const float*)d_in[1];
  const float* nang = (const float*)d_in[2];
  const float* Wq = (const float*)d_in[3];   const float* bq = (const float*)d_in[4];
  const float* Wk = (const float*)d_in[5];   const float* bk = (const float*)d_in[6];
  const float* Wv = (const float*)d_in[7];   const float* bv = (const float*)d_in[8];
  const float* Wke = (const float*)d_in[9];  const float* bke = (const float*)d_in[10];
  const float* Wve = (const float*)d_in[11]; const float* bve = (const float*)d_in[12];
  const float* We = (const float*)d_in[13];
  const float* Wc = (const float*)d_in[14];  const float* bc = (const float*)d_in[15];
  const float* Wk1 = (const float*)d_in[16]; const float* bk1 = (const float*)d_in[17];
  const float* Wk2 = (const float*)d_in[18]; const float* bk2 = (const float*)d_in[19];
  const float* Wm1 = (const float*)d_in[20]; const float* bm1 = (const float*)d_in[21];
  const float* Wm2 = (const float*)d_in[22]; const float* bm2 = (const float*)d_in[23];
  const float* g_att = (const float*)d_in[24]; const float* b_att = (const float*)d_in[25];
  const float* g_bn  = (const float*)d_in[26]; const float* b_bn  = (const float*)d_in[27];
  float* out = (float*)d_out;
  const int E = in_sizes[0] / 128;

  bf16_t* wsb = (bf16_t*)d_ws;
  auto cvt = [&](const float* W, int off, int K, int N) {
    int n = K * N;
    cvt_transpose_bf16<<<(n + 255) / 256, 256, 0, stream>>>(W, wsb + off, K, N);
  };
  cvt(Wq, OFF_WQ, 128, 128);
  cvt(Wk, OFF_WK, 128, 128);
  cvt(Wv, OFF_WV, 128, 128);
  for (int t = 0; t < 3; ++t) cvt(Wke + t * 16384, OFF_WKE + t * 16384, 128, 128);
  for (int t = 0; t < 3; ++t) cvt(Wve + t * 16384, OFF_WVE + t * 16384, 128, 128);
  cvt(We,  OFF_WE,  128, 128);
  cvt(Wk2, OFF_WK2, 128, 128);
  cvt(Wm2, OFF_WM2, 128, 128);
  cvt(Wc,  OFF_WC,  128, 128);
  cvt(Wk1, OFF_WK1, 384, 128);
  cvt(Wm1, OFF_WM1, 384, 128);

  int grid = (E + BE - 1) / BE;
  comformer_edge_kernel<<<grid, 256, 0, stream>>>(
      edge, nlen, nang, bq, bk, bv, bke, bve, bk1, bk2, bm1, bm2, bc,
      g_att, b_att, g_bn, b_bn, wsb, out, E);
}
